// HD95Loss_63488206569637
// MI455X (gfx1250) — matrix operand order repaired and checked by reading, not compiled, not run on hardware
//
#include <hip/hip_runtime.h>
#include <hip/hip_bf16.h>

typedef __attribute__((ext_vector_type(16))) _Float16 v16h;
typedef __attribute__((ext_vector_type(8)))  float    v8f;

#define HH 256
#define WW 256
#define NPIX 65536
#define NIMG 4
#define NBINS 130052          // squared distances 0..130050 + overflow bin
#define OVERFLOW_BIN 130051
#define BIG2F 1.0e12f         // BIG*BIG filler used by reference percentile

// ---------------------------------------------------------------------------
// Boundary extraction via WMMA: cross = T*M + M*T - M  (T = tridiagonal ones)
// cross == 5  <=>  pixel and its 4 cross neighbors all set  <=>  erode() true
// boundary = m && (cross < 5). Exact: sums <=5 in f16->f32 WMMA accumulation.
// blockIdx.x = 16x16 output tile (bi*16+bj), .y = image, .z = 0:pred 1:target
// ---------------------------------------------------------------------------
__global__ __launch_bounds__(32)
void boundary_wmma_kernel(const float* __restrict__ pred,
                          const int*   __restrict__ target,
                          unsigned char* __restrict__ ep,
                          unsigned char* __restrict__ eg) {
    __shared__ _Float16 lds_col[HH * 16];  // M[0:256, 16bj:16bj+16]
    __shared__ _Float16 lds_row[16 * WW];  // M[16bi:16bi+16, 0:256]
    const int tile = blockIdx.x;
    const int bi = tile >> 4, bj = tile & 15;
    const int b = blockIdx.y;
    const int which = blockIdx.z;          // 0: pm from pred, 1: gm from target
    const int l = threadIdx.x;             // 0..31 (one wave)
    const float* pbase = pred + b * NPIX;
    const int*   tbase = target + b * NPIX;

    for (int e = l; e < HH * 16; e += 32) {
        int k = e >> 4, n = e & 15;
        int g = k * WW + (bj * 16 + n);
        float mv = which ? (tbase[g] > 0 ? 1.f : 0.f) : (pbase[g] > 0.f ? 1.f : 0.f);
        lds_col[e] = (_Float16)mv;
    }
    for (int e = l; e < 16 * WW; e += 32) {
        int r = e >> 8, c = e & 255;
        int g = (bi * 16 + r) * WW + c;
        float mv = which ? (tbase[g] > 0 ? 1.f : 0.f) : (pbase[g] > 0.f ? 1.f : 0.f);
        lds_row[e] = (_Float16)mv;
    }
    __syncthreads();

    const int half = l >> 4;               // lane half selects K sub-range
    const int lm   = l & 15;
    v8f acc = {};                          // accumulates T*M + M*T for this tile

    for (int k0 = 0; k0 < HH; k0 += 32) {
        v16h a1, b1, a2, b2;
        #pragma unroll
        for (int v = 0; v < 8; ++v) {
            // A layout (16x32 f16): lane half/group select K (ISA 7.12.2)
            int K0 = (v >> 2) * 16 + half * 8 + (v & 3) * 2;
            int i  = bi * 16 + lm;         // global output row
            int j0 = k0 + K0, j1 = j0 + 1; // global K (contraction) indices
            // A1 = T tile (procedural tridiagonal)
            a1[2 * v]     = (_Float16)(((i - j0 <= 1) && (j0 - i <= 1)) ? 1.f : 0.f);
            a1[2 * v + 1] = (_Float16)(((i - j1 <= 1) && (j1 - i <= 1)) ? 1.f : 0.f);
            // A2 = M row-block tile
            a2[2 * v]     = lds_row[lm * WW + j0];
            a2[2 * v + 1] = lds_row[lm * WW + j1];
            // B layout (32x16 f16): col N = lane%16, K same as A (K0)
            int kb0 = k0 + K0, kb1 = kb0 + 1;
            b1[2 * v]     = lds_col[kb0 * 16 + lm];
            b1[2 * v + 1] = lds_col[kb1 * 16 + lm];
            int wc = bj * 16 + lm;         // global output col
            // B2 = T tile (procedural)
            b2[2 * v]     = (_Float16)(((kb0 - wc <= 1) && (wc - kb0 <= 1)) ? 1.f : 0.f);
            b2[2 * v + 1] = (_Float16)(((kb1 - wc <= 1) && (wc - kb1 <= 1)) ? 1.f : 0.f);
        }
        acc = __builtin_amdgcn_wmma_f32_16x16x32_f16(false, a1, false, b1,
                                                     (short)0, acc, false, false);
        acc = __builtin_amdgcn_wmma_f32_16x16x32_f16(false, a2, false, b2,
                                                     (short)0, acc, false, false);
    }

    unsigned char* outp = which ? eg : ep;
    #pragma unroll
    for (int v = 0; v < 8; ++v) {
        int rloc = v + half * 8;           // C layout: M = v + 8*(lane/16)
        int row = bi * 16 + rloc;
        int col = bj * 16 + lm;            // N = lane%16
        float m = (float)lds_row[rloc * WW + col];
        float cross = acc[v] - m;          // up+down+left+right+center
        // cross includes center twice (once per product) minus once here -> 5-sum
        bool boundary = (m > 0.5f) && (cross < 4.5f);
        outp[b * NPIX + row * WW + col] = boundary ? (unsigned char)1 : (unsigned char)0;
    }
}

// ---------------------------------------------------------------------------
// 1D row distance (capped at 1e6), squared. One block per (row, image).
// ---------------------------------------------------------------------------
__global__ __launch_bounds__(256)
void rowdist_kernel(const unsigned char* __restrict__ mask,
                    float* __restrict__ d1sq) {
    __shared__ int flags[WW];
    int b = blockIdx.y, row = blockIdx.x, w = threadIdx.x;
    int base = b * NPIX + row * WW;
    flags[w] = mask[base + w];
    __syncthreads();
    int best = 1 << 30;
    for (int w2 = 0; w2 < WW; ++w2) {
        if (flags[w2]) { int d = w - w2; if (d < 0) d = -d; if (d < best) best = d; }
    }
    float d = (best > 1000000) ? 1.0e6f : (float)best;
    d1sq[base + w] = d * d;                // empty row -> fl(1e12), matches ref
}

// ---------------------------------------------------------------------------
// Column pass: d2sq[i,w] = min_j (d1sq[j,w] + (i-j)^2). Exact in fp32.
// ---------------------------------------------------------------------------
__global__ __launch_bounds__(256)
void colpass_kernel(const float* __restrict__ d1sq,
                    float* __restrict__ d2sq) {
    __shared__ float col[HH];
    int b = blockIdx.y, w = blockIdx.x, i = threadIdx.x;
    col[i] = d1sq[b * NPIX + i * WW + w];
    __syncthreads();
    float best = 3.0e38f;
    for (int j = 0; j < HH; ++j) {
        int dj = i - j;
        best = fminf(best, col[j] + (float)(dj * dj));
    }
    d2sq[b * NPIX + i * WW + w] = best;
}

// ---------------------------------------------------------------------------
// Histogram of integer squared distances at masked pixels (per image).
// ---------------------------------------------------------------------------
__global__ __launch_bounds__(256)
void hist_kernel(const float* __restrict__ d2sq,
                 const unsigned char* __restrict__ mask,
                 unsigned int* __restrict__ hist) {
    int idx = blockIdx.x * 256 + threadIdx.x;
    if (idx >= NIMG * NPIX) return;
    if (!mask[idx]) return;
    int b = idx >> 16;
    float v = d2sq[idx];
    int bin = (v < 130050.5f) ? (int)(v + 0.5f) : OVERFLOW_BIN;
    atomicAdd(&hist[b * NBINS + bin], 1u);
}

// ---------------------------------------------------------------------------
// Exact 95th percentile with linear interpolation via histogram selection.
// One block per image. Ranks >= n take the BIG*BIG (1e12) filler like the ref.
// ---------------------------------------------------------------------------
__global__ __launch_bounds__(256)
void select_kernel(const unsigned int* __restrict__ hist,
                   float* __restrict__ results, int roff) {
    __shared__ unsigned int csum[256];
    __shared__ unsigned int cpref[256];
    __shared__ int sn;
    __shared__ float svals[2];
    const int CH = (NBINS + 255) / 256;    // 509 bins per thread chunk
    int b = blockIdx.x, t = threadIdx.x;
    const unsigned int* hb = hist + b * NBINS;

    unsigned int s = 0;
    for (int i = 0; i < CH; ++i) {
        int g = t * CH + i;
        if (g < NBINS) s += hb[g];
    }
    csum[t] = s;
    if (t < 2) svals[t] = BIG2F;           // default: rank beyond masked count
    __syncthreads();
    if (t == 0) {
        unsigned int run = 0;
        for (int u = 0; u < 256; ++u) { cpref[u] = run; run += csum[u]; }
        sn = (int)run;
    }
    __syncthreads();

    int n = sn;
    float pos = 0.95f * (float)(n > 1 ? n - 1 : 0);
    int lo = (int)floorf(pos);
    if (lo < 0) lo = 0; if (lo > NPIX - 1) lo = NPIX - 1;
    int hi = lo + 1; if (hi > NPIX - 1) hi = NPIX - 1;
    float frac = pos - floorf(pos);

    #pragma unroll
    for (int ri = 0; ri < 2; ++ri) {
        int r = (ri == 0) ? lo : hi;
        if (r < n && (unsigned)r >= cpref[t] && (unsigned)r < cpref[t] + csum[t]) {
            unsigned int c = cpref[t];
            for (int i = 0; i < CH; ++i) {
                int g = t * CH + i;
                if (g >= NBINS) break;
                unsigned int hv = hb[g];
                if ((unsigned)r < c + hv) {
                    float sq = (g == OVERFLOW_BIN) ? BIG2F : (float)g;
                    svals[ri] = sqrtf(sq);
                    break;
                }
                c += hv;
            }
        }
    }
    __syncthreads();
    if (t == 0)
        results[roff + b] = svals[0] * (1.0f - frac) + svals[1] * frac;
}

__global__ void final_kernel(const float* __restrict__ results,
                             float* __restrict__ out) {
    if (threadIdx.x == 0 && blockIdx.x == 0) {
        float s = 0.f;
        for (int b = 0; b < NIMG; ++b)
            s += fmaxf(results[b], results[4 + b]);
        out[0] = s * 0.25f;
    }
}

// ---------------------------------------------------------------------------
extern "C" void kernel_launch(void* const* d_in, const int* in_sizes, int n_in,
                              void* d_out, int out_size, void* d_ws, size_t ws_size,
                              hipStream_t stream) {
    (void)in_sizes; (void)n_in; (void)out_size; (void)ws_size;
    const float* pred  = (const float*)d_in[0];
    const int*   target = (const int*)d_in[1];
    float* out = (float*)d_out;
    char* ws = (char*)d_ws;

    unsigned char* ep   = (unsigned char*)(ws + 0);        // 262144 B
    unsigned char* eg   = (unsigned char*)(ws + 262144);   // 262144 B
    float*         d1sq = (float*)(ws + 524288);           // 1 MiB
    float*         d2sq = (float*)(ws + 1572864);          // 1 MiB
    unsigned int*  hist = (unsigned int*)(ws + 2621440);   // 2080832 B
    float*         results = (float*)(ws + 4702272);       // 32 B

    // boundaries for both masks (WMMA cross-erosion)
    boundary_wmma_kernel<<<dim3(256, NIMG, 2), 32, 0, stream>>>(pred, target, ep, eg);

    // direction 0: d_pg = pct95( EDT(eg) at ep )
    rowdist_kernel<<<dim3(HH, NIMG), 256, 0, stream>>>(eg, d1sq);
    colpass_kernel<<<dim3(WW, NIMG), 256, 0, stream>>>(d1sq, d2sq);
    hipMemsetAsync(hist, 0, NIMG * NBINS * sizeof(unsigned int), stream);
    hist_kernel<<<dim3((NIMG * NPIX) / 256), 256, 0, stream>>>(d2sq, ep, hist);
    select_kernel<<<dim3(NIMG), 256, 0, stream>>>(hist, results, 0);

    // direction 1: d_gp = pct95( EDT(ep) at eg )
    rowdist_kernel<<<dim3(HH, NIMG), 256, 0, stream>>>(ep, d1sq);
    colpass_kernel<<<dim3(WW, NIMG), 256, 0, stream>>>(d1sq, d2sq);
    hipMemsetAsync(hist, 0, NIMG * NBINS * sizeof(unsigned int), stream);
    hist_kernel<<<dim3((NIMG * NPIX) / 256), 256, 0, stream>>>(d2sq, eg, hist);
    select_kernel<<<dim3(NIMG), 256, 0, stream>>>(hist, results, 4);

    final_kernel<<<1, 1, 0, stream>>>(results, out);
}
